// BaselineMIPSModule_8744553415357
// MI455X (gfx1250) — matrix operand order repaired and checked by reading, not compiled, not run on hardware
//
#include <hip/hip_runtime.h>

typedef unsigned short ushort_t;
typedef unsigned int   uint_t;
typedef __attribute__((ext_vector_type(16))) __bf16         v16bf;
typedef __attribute__((ext_vector_type(8)))  float          v8f;
typedef __attribute__((ext_vector_type(8)))  unsigned short ushort8;
typedef __attribute__((ext_vector_type(16))) unsigned short ushort16;
typedef __attribute__((ext_vector_type(4)))  float          f4;

#define N_CORPUS 500000
#define DIM      128
#define NQ       512
#define KTOP     100
#define SLAB     4096
#define NSLAB    123                 // ceil(500000/4096); 123*4096 = 503808
#define NPAD     (NSLAB * SLAB)
#define QT       16
#define NQG      (NQ / QT)           // 32
#define TILES    (SLAB / 16)         // 256
#define NITER    (TILES / 8)         // 32 tiles per wave (uniform)
#define NCAND    (NSLAB * KTOP)      // 12300
#define NEG_INF  (-__builtin_inff())

__device__ __forceinline__ ushort_t f32_to_bf16(float f) {
  uint_t u = __builtin_bit_cast(uint_t, f);
  u += 0x7FFFu + ((u >> 16) & 1u);   // round-to-nearest-even
  return (ushort_t)(u >> 16);
}

// Concatenate two 8x16-bit vectors into one 16x bf16 WMMA fragment without
// ever touching scalar __bf16 (keeps IR as load/shuffle/bitcast so the
// backend feeds the fragment registers directly from b128 loads).
__device__ __forceinline__ v16bf concat_bf16(ushort8 a, ushort8 b) {
  ushort16 w = __builtin_shufflevector(a, b, 0, 1, 2, 3, 4, 5, 6, 7,
                                             8, 9, 10, 11, 12, 13, 14, 15);
  return __builtin_bit_cast(v16bf, w);
}

// Load one 16x128 corpus tile's A fragments (4 K-chunks, 2x b128 each).
__device__ __forceinline__ void load_afrag(v16bf a[4], const ushort_t* rp) {
#pragma unroll
  for (int c = 0; c < 4; ++c) {
    ushort8 x0 = *(const ushort8*)(rp + c * 32);       // K0..K0+7
    ushort8 x1 = *(const ushort8*)(rp + c * 32 + 16);  // K0+16..K0+23
    a[c] = concat_bf16(x0, x1);
  }
}

// 4 WMMAs + masked score store for one tile.
__device__ __forceinline__ void wmma_tile(const v16bf a[4], const v16bf b[4],
                                          float* __restrict__ scores,
                                          int lq, int lhalf, int t,
                                          size_t slab_row0) {
  v8f acc = {};
#pragma unroll
  for (int c = 0; c < 4; ++c)
    acc = __builtin_amdgcn_wmma_f32_16x16x32_bf16(
        false, a[c], false, b[c], (short)0, acc, false, false);

  // D layout: VGPR r, lanes 0-15 -> M=r, lanes 16-31 -> M=r+8; N = lane%16.
  const int rbase = t * 16 + lhalf * 8;
  float tmp[8];
#pragma unroll
  for (int r = 0; r < 8; ++r) {
    size_t grow = slab_row0 + rbase + r;
    tmp[r] = (grow < (size_t)N_CORPUS) ? acc[r] : NEG_INF;  // mask pad rows
  }
  f4* dst = (f4*)&scores[lq * SLAB + rbase];
  f4 lo, hi;
  lo[0]=tmp[0]; lo[1]=tmp[1]; lo[2]=tmp[2]; lo[3]=tmp[3];
  hi[0]=tmp[4]; hi[1]=tmp[5]; hi[2]=tmp[6]; hi[3]=tmp[7];
  dst[0] = lo; dst[1] = hi;
}

// ---------------------------------------------------------------- pass 0 ----
// fp32 corpus -> bf16 (padded rows zero-filled), packed 2 elems per u32 store.
__global__ void convert_corpus_bf16(const float* __restrict__ corpus,
                                    ushort_t* __restrict__ out) {
  const long long pairs  = (long long)NPAD * DIM / 2;
  const long long nreal  = (long long)N_CORPUS * DIM;
  const long long stride = (long long)gridDim.x * blockDim.x;
  for (long long i = (long long)blockIdx.x * blockDim.x + threadIdx.x;
       i < pairs; i += stride) {
    long long e = i * 2;
    uint_t lo = 0, hi = 0;
    if (e < nreal) {               // row length (128) is even -> e+1 also real
      lo = f32_to_bf16(corpus[e]);
      hi = f32_to_bf16(corpus[e + 1]);
    }
    ((uint_t*)out)[i] = lo | (hi << 16);
  }
}

// ---------------------------------------------------------------- pass 1 ----
// Fused bf16 WMMA GEMM (scores stay in LDS) + per-slab top-100 per query.
// D = corpusTile(16 x K) * Q^T(K x 16); M = corpus row, N = query.
// A fragments are double-buffered in registers (ping-pong, unroll-by-2) so
// the next tile's 8x global_load_b128 issue before the current tile's WMMAs.
__global__ void __launch_bounds__(256)
mips_pass1(const float* __restrict__ query,
           const ushort_t* __restrict__ corpus_bf,
           float* __restrict__ cand_s, int* __restrict__ cand_i) {
  extern __shared__ char smem[];
  ushort_t* q_sh   = (ushort_t*)smem;                 // QT*DIM bf16 (4 KB)
  float*    scores = (float*)(smem + QT * DIM * 2);   // QT*SLAB f32 (256 KB)

  const int tid   = threadIdx.x;
  const int qg    = blockIdx.x;           // query group 0..31
  const int slab  = blockIdx.y;           // 0..122
  const int qbase = qg * QT;

  // Stage + convert 16 query rows into LDS (bf16, row-major).
  for (int i = tid; i < QT * DIM; i += 256)
    q_sh[i] = f32_to_bf16(query[(size_t)qbase * DIM + i]);
  __syncthreads();

  const int wave  = tid >> 5;
  const int lane  = tid & 31;
  const int lhalf = (lane < 16) ? 0 : 1;
  const int lq    = lane & 15;

  // B fragments (Q^T, 32x16 per chunk): lane n<16 holds K {0-7,16-23}, n>=16 K {8-15,24-31}.
  v16bf bfrag[4];
#pragma unroll
  for (int c = 0; c < 4; ++c) {
    const ushort8* src = (const ushort8*)&q_sh[lq * DIM + c * 32 + lhalf * 8];
    bfrag[c] = concat_bf16(src[0], src[2]);
  }

  const size_t slab_row0 = (size_t)slab * SLAB;
  const size_t tstride   = (size_t)8 * 16 * DIM;        // elems per wave step

  // A fragments: lane m holds K {0-7,16-23}, lane m+16 holds K {8-15,24-31}.
  const ushort_t* rp =
      corpus_bf + (slab_row0 + (size_t)wave * 16 + lq) * DIM + lhalf * 8;

  v16bf a0[4], a1[4];
  load_afrag(a0, rp);

  for (int i = 0; i < NITER; i += 2) {                  // NITER==32, even
    const int t0 = wave + 8 * i;
    const int t1 = t0 + 8;
    __builtin_prefetch(rp + 3 * tstride, 0, 1);         // global_prefetch_b8
    load_afrag(a1, rp + tstride);                       // tile i+1 (always valid)
    wmma_tile(a0, bfrag, scores, lq, lhalf, t0, slab_row0);
    if (i + 2 < NITER)                                  // uniform scalar guard
      load_afrag(a0, rp + 2 * tstride);                 // tile i+2
    wmma_tile(a1, bfrag, scores, lq, lhalf, t1, slab_row0);
    rp += 2 * tstride;
  }
  __syncthreads();

  // Per-slab top-100: wave w owns queries 2w (lanes 0-15) and 2w+1 (lanes 16-31).
  const int qi = 2 * wave + lhalf;
  const int s  = lq;
  float* qs = &scores[qi * SLAB];
  for (int it = 0; it < KTOP; ++it) {
    float bv = NEG_INF; int bp = 0;
    for (int j = s; j < SLAB; j += 16) {
      float v = qs[j];
      if (v > bv) { bv = v; bp = j; }
    }
#pragma unroll
    for (int m = 8; m >= 1; m >>= 1) {           // stays inside the half-wave
      float ov = __shfl_xor(bv, m, 32);
      int   op = __shfl_xor(bp, m, 32);
      if (ov > bv) { bv = ov; bp = op; }
    }
    if (s == 0) {
      int o = ((qbase + qi) * NSLAB + slab) * KTOP + it;
      cand_s[o] = bv;
      cand_i[o] = (int)(slab_row0 + bp);
      qs[bp] = NEG_INF;
    }
    __syncthreads();
  }
}

// ---------------------------------------------------------------- pass 2 ----
// Merge 123*100 candidates per query -> final top-100 (descending) + gather.
__global__ void __launch_bounds__(256)
mips_pass2(const float* __restrict__ corpus,
           const float* __restrict__ cand_s, const int* __restrict__ cand_i,
           int* __restrict__ idx_out, float* __restrict__ sc_out,
           float* __restrict__ emb_out) {
  extern __shared__ char smem[];
  float* cs   = (float*)smem;                              // NCAND
  int*   ci   = (int*)(smem + (size_t)NCAND * 4);          // NCAND
  int*   widx = (int*)(smem + (size_t)NCAND * 8);          // KTOP
  float* redv = (float*)(smem + (size_t)NCAND * 8 + KTOP * 4);       // 8
  int*   redp = (int*)(smem + (size_t)NCAND * 8 + KTOP * 4 + 32);    // 8

  const int q    = blockIdx.x;
  const int tid  = threadIdx.x;
  const int wave = tid >> 5, lane = tid & 31;

  for (int i = tid; i < NCAND; i += 256) {
    cs[i] = cand_s[(size_t)q * NCAND + i];
    ci[i] = cand_i[(size_t)q * NCAND + i];
  }
  __syncthreads();

  for (int it = 0; it < KTOP; ++it) {
    float bv = NEG_INF; int bp = 0;
    for (int j = tid; j < NCAND; j += 256) {
      float v = cs[j];
      if (v > bv) { bv = v; bp = j; }
    }
#pragma unroll
    for (int m = 16; m >= 1; m >>= 1) {
      float ov = __shfl_xor(bv, m, 32);
      int   op = __shfl_xor(bp, m, 32);
      if (ov > bv) { bv = ov; bp = op; }
    }
    if (lane == 0) { redv[wave] = bv; redp[wave] = bp; }
    __syncthreads();
    if (tid == 0) {
      float fv = redv[0]; int fp = redp[0];
      for (int w = 1; w < 8; ++w)
        if (redv[w] > fv) { fv = redv[w]; fp = redp[w]; }
      int idx = ci[fp];
      if (idx < 0 || idx >= N_CORPUS) idx = 0;
      idx_out[q * KTOP + it] = idx;
      sc_out[q * KTOP + it]  = fv;
      widx[it] = idx;
      cs[fp] = NEG_INF;
    }
    __syncthreads();
  }

  // Gather embeddings: one row per wave, float4 per lane (32*4 = 128 floats).
  for (int i = wave; i < KTOP; i += 8) {
    int idx = widx[i];
    const f4* src = (const f4*)(corpus + (size_t)idx * DIM);
    f4*       dst = (f4*)(emb_out + ((size_t)q * KTOP + i) * DIM);
    dst[lane] = src[lane];
  }
}

// -------------------------------------------------------------- launcher ----
extern "C" void kernel_launch(void* const* d_in, const int* in_sizes, int n_in,
                              void* d_out, int out_size, void* d_ws, size_t ws_size,
                              hipStream_t stream) {
  const float* query  = (const float*)d_in[0];   // [512,128] f32
  const float* corpus = (const float*)d_in[1];   // [500000,128] f32
  // d_in[2] = num_items (==100, baked in as KTOP)

  // Workspace layout (~179.4 MB total):
  //   [0)                corpus_bf16: NPAD*128 bf16 = 128,974,848 B
  //   [o_cbf)            cand_s: 512*123*100 f32    =  25,190,400 B
  //   [o_cbf+cand_bytes) cand_i: 512*123*100 i32    =  25,190,400 B
  char* ws = (char*)d_ws;
  ushort_t* corpus_bf  = (ushort_t*)ws;
  size_t    o_cbf      = (size_t)NPAD * DIM * 2;
  size_t    cand_bytes = (size_t)NQ * NSLAB * KTOP * 4;
  float*    cand_s     = (float*)(ws + o_cbf);
  int*      cand_i     = (int*)(ws + o_cbf + cand_bytes);

  // d_out: [indices i32 512*100][scores f32 512*100][embeddings f32 512*100*128]
  int*   idx_out = (int*)d_out;
  float* sc_out  = (float*)d_out + (size_t)NQ * KTOP;
  float* emb_out = (float*)d_out + 2 * (size_t)NQ * KTOP;

  convert_corpus_bf16<<<2048, 256, 0, stream>>>(corpus, corpus_bf);

  dim3 g1(NQG, NSLAB);
  size_t lds1 = (size_t)QT * DIM * 2 + (size_t)QT * SLAB * 4;   // 266,240 B
  mips_pass1<<<g1, 256, lds1, stream>>>(query, corpus_bf, cand_s, cand_i);

  size_t lds2 = (size_t)NCAND * 8 + KTOP * 4 + 64;              // 98,864 B
  mips_pass2<<<NQ, 256, lds2, stream>>>(corpus, cand_s, cand_i,
                                        idx_out, sc_out, emb_out);
}
